// SparseDistStack_49271864819945
// MI455X (gfx1250) — compile-verified, matrix operand order checked by reading, not run on hardware
//
#include <hip/hip_runtime.h>
#include <hip/hip_bf16.h>

typedef __attribute__((ext_vector_type(16))) _Float16 v16h;
typedef __attribute__((ext_vector_type(8)))  float    v8f;

#define WMMA_F32_F16(A, B, C) \
    __builtin_amdgcn_wmma_f32_16x16x32_f16(false, (A), false, (B), (short)0, (C), false, false)

__device__ __forceinline__ float sigmoidf_(float x) { return 1.0f / (1.0f + __expf(-x)); }

// Fragment load/store as 2 x 16B so the compiler emits ds_load_b128/ds_store_b128.
__device__ __forceinline__ v16h ldFrag(const _Float16* p) {
    union { uint4 u[2]; v16h h; } c;
    const uint4* q = (const uint4*)p;
    c.u[0] = q[0]; c.u[1] = q[1];
    return c.h;
}
__device__ __forceinline__ void stFrag(_Float16* p, v16h v) {
    union { uint4 u[2]; v16h h; } c; c.h = v;
    uint4* q = (uint4*)p;
    q[0] = c.u[0]; q[1] = c.u[1];
}

// Swizzle maps (ISA 7.12.2, wave32 16-bit A/B fragments):
//   lane = (n % 16) + 16 * ((k_local % 16) >= 8)
//   idx  = 8 * (k_local >= 16) + (k_local % 8)
// We store LDS tiles as [32 lanes][16 halves] so each lane reads 32B contiguous.

// ---------------------------------------------------------------------------
// Kernel 1: tri-update "pre" pass: zn = LN(z); fused GEMM [16,64]x[64,128]
//   cols 0-15 ga | 16-31 la | 32-47 gb | 48-63 lb | 64-127 go
//   a = sig(ga)*la, b = sig(gb)*lb (f16 [E,16]); go = sig(go) (f16 [E,64]).
// 8 waves/block, one 16-edge M-tile per wave.
// ---------------------------------------------------------------------------
__global__ __launch_bounds__(256) void k_ab(
    const float* __restrict__ z,
    const float* __restrict__ ln_g, const float* __restrict__ ln_b,
    const float* __restrict__ gaW, const float* __restrict__ gab,
    const float* __restrict__ laW, const float* __restrict__ lab,
    const float* __restrict__ gbW, const float* __restrict__ gbb,
    const float* __restrict__ lbW, const float* __restrict__ lbb,
    const float* __restrict__ goW, const float* __restrict__ gob,
    _Float16* __restrict__ a_out, _Float16* __restrict__ b_out,
    _Float16* __restrict__ go_out, int numTiles)
{
    __shared__ __align__(16) _Float16 Wsw[2][8][32][16];   // [ks][nt][lane][idx] 16KB
    __shared__ __align__(16) _Float16 Asw[8][2][32][16];   // [wave][ks][lane][idx] 16KB
    __shared__ float bias[128];

    const int tid = threadIdx.x;

    // Stage combined weights, pre-swizzled. 2*8*32 = 512 lane-units.
    for (int u = tid; u < 512; u += 256) {
        int lane_u = u & 31;
        int tileId = u >> 5;          // 0..15
        int ks = tileId >> 3, nt = tileId & 7;
        int n  = nt * 16 + (lane_u & 15);
        int kb = (lane_u >= 16) ? 8 : 0;
        const float* src; int stride;
        if      (n < 16)  { src = gaW + n;        stride = 16; }
        else if (n < 32)  { src = laW + (n - 16); stride = 16; }
        else if (n < 48)  { src = gbW + (n - 32); stride = 16; }
        else if (n < 64)  { src = lbW + (n - 48); stride = 16; }
        else              { src = goW + (n - 64); stride = 64; }
        v16h w;
        #pragma unroll
        for (int idx = 0; idx < 16; ++idx) {
            int k = ks * 32 + ((idx >> 3) * 16) + kb + (idx & 7);
            w[idx] = (_Float16)src[k * stride];
        }
        stFrag(&Wsw[ks][nt][lane_u][0], w);
    }
    if (tid < 128) {
        float bv;
        if      (tid < 16) bv = gab[tid];
        else if (tid < 32) bv = lab[tid - 16];
        else if (tid < 48) bv = gbb[tid - 32];
        else if (tid < 64) bv = lbb[tid - 48];
        else               bv = gob[tid - 64];
        bias[tid] = bv;
    }

    const int wave = tid >> 5, lane = tid & 31;
    const int tile = blockIdx.x * 8 + wave;
    const bool active = (tile < numTiles);

    if (active && lane < 16) {
        long e = (long)tile * 16 + lane;
        const float* zr = z + e * 64;
        float v[64], s1 = 0.f, s2 = 0.f;
        #pragma unroll
        for (int c = 0; c < 64; ++c) { float x = zr[c]; v[c] = x; s1 += x; s2 += x * x; }
        float mean = s1 * (1.0f / 64.0f);
        float var  = s2 * (1.0f / 64.0f) - mean * mean;
        float inv  = rsqrtf(var + 1e-5f);
        #pragma unroll
        for (int c = 0; c < 64; ++c) v[c] = (v[c] - mean) * inv * ln_g[c] + ln_b[c];
        // Row r owns 4 contiguous fragment chunks: (ks, lane r) and (ks, lane r+16).
        #pragma unroll
        for (int ks = 0; ks < 2; ++ks) {
            v16h t0, t1;
            #pragma unroll
            for (int idx = 0; idx < 16; ++idx) {
                int kl = ((idx >> 3) * 16) + (idx & 7);   // {0..7, 16..23}
                t0[idx] = (_Float16)v[ks * 32 + kl];
                t1[idx] = (_Float16)v[ks * 32 + 8 + kl];  // {8..15, 24..31}
            }
            stFrag(&Asw[wave][ks][lane][0], t0);
            stFrag(&Asw[wave][ks][lane + 16][0], t1);
        }
    }
    __syncthreads();
    if (!active) return;

    const int nl   = lane & 15;
    const int moff = (lane >= 16) ? 8 : 0;

    v16h a0 = ldFrag(&Asw[wave][0][lane][0]);
    v16h a1 = ldFrag(&Asw[wave][1][lane][0]);

    v8f accv[8];
    #pragma unroll
    for (int nt = 0; nt < 8; ++nt) {
        v16h b0 = ldFrag(&Wsw[0][nt][lane][0]);
        v16h b1 = ldFrag(&Wsw[1][nt][lane][0]);
        v8f c = {};
        c = WMMA_F32_F16(a0, b0, c);
        c = WMMA_F32_F16(a1, b1, c);
        accv[nt] = c;
    }

    const long ebase = (long)tile * 16;
    #pragma unroll
    for (int r = 0; r < 8; ++r) {
        long e = ebase + r + moff;
        float ga = accv[0][r] + bias[nl];
        float la = accv[1][r] + bias[16 + nl];
        a_out[e * 16 + nl] = (_Float16)(sigmoidf_(ga) * la);
        float gb = accv[2][r] + bias[32 + nl];
        float lb = accv[3][r] + bias[48 + nl];
        b_out[e * 16 + nl] = (_Float16)(sigmoidf_(gb) * lb);
        #pragma unroll
        for (int q = 0; q < 4; ++q) {
            int col = q * 16 + nl;
            go_out[e * 64 + col] = (_Float16)sigmoidf_(accv[4 + q][r] + bias[64 + col]);
        }
    }
}

// ---------------------------------------------------------------------------
// Kernel 2: kNN gather + per-channel K=12 contraction + LN(16) + 16->64 + gate.
//   row: k[i,m,c] = sum_t a[i,t,c] * b[idx[i,m],t,c]
//   col: k[i,m,c] = sum_t a[idx[i,m],t,c] * b[i,t,c]
// ---------------------------------------------------------------------------
__global__ __launch_bounds__(256) void k_gather(
    const _Float16* __restrict__ a_buf, const _Float16* __restrict__ b_buf,
    const _Float16* __restrict__ go_buf,
    const int* __restrict__ nbr,
    const float* __restrict__ lno_g, const float* __restrict__ lno_b,
    const float* __restrict__ loW, const float* __restrict__ lob,
    float* __restrict__ acc, int isCol, int numEdges)
{
    __shared__ float Wo[16][64];
    __shared__ float bo[64];
    const int tid = threadIdx.x;
    for (int t = tid; t < 16 * 64; t += 256) Wo[t >> 6][t & 63] = loW[t];
    if (tid < 64) bo[tid] = lob[tid];
    __syncthreads();

    const int  eLocal = tid >> 4;
    const int  c      = tid & 15;
    const long e      = (long)blockIdx.x * 16 + eLocal;
    if (e >= numEdges) return;
    const int i  = (int)(e / 12);
    const int mm = (int)(e % 12);
    const int j  = nbr[i * 12 + mm];

    const long abase = (long)(isCol ? j : i) * 12 * 16;
    const long bbase = (long)(isCol ? i : j) * 12 * 16;
    float kval = 0.f;
    #pragma unroll
    for (int t = 0; t < 12; ++t)
        kval += (float)a_buf[abase + t * 16 + c] * (float)b_buf[bbase + t * 16 + c];

    float s1 = kval, s2 = kval * kval;
    #pragma unroll
    for (int msk = 1; msk < 16; msk <<= 1) {
        s1 += __shfl_xor(s1, msk, 32);
        s2 += __shfl_xor(s2, msk, 32);
    }
    float mean = s1 * (1.f / 16.f);
    float var  = s2 * (1.f / 16.f) - mean * mean;
    float kn   = (kval - mean) * rsqrtf(var + 1e-5f) * lno_g[c] + lno_b[c];

    float o0 = bo[c], o1 = bo[c + 16], o2 = bo[c + 32], o3 = bo[c + 48];
    const int gbase = (tid & 31) & 16;
    #pragma unroll
    for (int cc = 0; cc < 16; ++cc) {
        float knc = __shfl(kn, gbase + cc, 32);
        o0 += knc * Wo[cc][c];
        o1 += knc * Wo[cc][c + 16];
        o2 += knc * Wo[cc][c + 32];
        o3 += knc * Wo[cc][c + 48];
    }
    const long ob = e * 64;
    float g0 = (float)go_buf[ob + c],      g1 = (float)go_buf[ob + c + 16];
    float g2 = (float)go_buf[ob + c + 32], g3 = (float)go_buf[ob + c + 48];
    if (isCol) {
        acc[ob + c]      += g0 * o0;  acc[ob + c + 16] += g1 * o1;
        acc[ob + c + 32] += g2 * o2;  acc[ob + c + 48] += g3 * o3;
    } else {
        acc[ob + c]      = g0 * o0;   acc[ob + c + 16] = g1 * o1;
        acc[ob + c + 32] = g2 * o2;   acc[ob + c + 48] = g3 * o3;
    }
}

// ---------------------------------------------------------------------------
// Kernel 3: residual + FFN. s = z + acc (residual stashed in acc),
//   h = gelu(LN(s)@W1+b1) [WMMA 64->256]; f = h@W2+b2 [WMMA 256->64]; z = s+f.
// 4 waves/block, one 16-edge tile per wave. Dynamic LDS (~105KB).
// ---------------------------------------------------------------------------
__global__ void k_ffn(
    float* __restrict__ z, float* __restrict__ acc,
    const float* __restrict__ ln_g, const float* __restrict__ ln_b,
    const float* __restrict__ W1, const float* __restrict__ b1,
    const float* __restrict__ W2, const float* __restrict__ b2,
    int numTiles)
{
    extern __shared__ char smem[];
    _Float16* W1sw = (_Float16*)smem;                 // [2][16][32][16]  32KB
    _Float16* W2sw = (_Float16*)(smem + 32768);       // [8][4][32][16]   32KB
    _Float16* Hsw  = (_Float16*)(smem + 65536);       // [4][8][32][16]   32KB
    _Float16* Ssw  = (_Float16*)(smem + 98304);       // [4][2][32][16]    8KB
    float*    b1s  = (float*)(smem + 106496);         // [256]
    float*    b2s  = (float*)(smem + 107520);         // [64]

    const int tid = threadIdx.x;

    // Stage W1 swizzled: 2*16*32 = 1024 lane-units.
    for (int u = tid; u < 1024; u += 128) {
        int lane_u = u & 31;
        int tileId = u >> 5;                  // 0..31
        int ks = tileId >> 4, nt = tileId & 15;
        int n  = nt * 16 + (lane_u & 15);
        int kb = (lane_u >= 16) ? 8 : 0;
        v16h w;
        #pragma unroll
        for (int idx = 0; idx < 16; ++idx) {
            int k = ks * 32 + ((idx >> 3) * 16) + kb + (idx & 7);
            w[idx] = (_Float16)W1[k * 256 + n];
        }
        stFrag(W1sw + ((ks * 16 + nt) * 32 + lane_u) * 16, w);
    }
    // Stage W2 swizzled: 8*4*32 = 1024 lane-units.
    for (int u = tid; u < 1024; u += 128) {
        int lane_u = u & 31;
        int tileId = u >> 5;                  // 0..31
        int ks = tileId >> 2, nt = tileId & 3;
        int n  = nt * 16 + (lane_u & 15);
        int kb = (lane_u >= 16) ? 8 : 0;
        v16h w;
        #pragma unroll
        for (int idx = 0; idx < 16; ++idx) {
            int k = ks * 32 + ((idx >> 3) * 16) + kb + (idx & 7);
            w[idx] = (_Float16)W2[k * 64 + n];
        }
        stFrag(W2sw + ((ks * 4 + nt) * 32 + lane_u) * 16, w);
    }
    for (int t = tid; t < 256; t += 128) b1s[t] = b1[t];
    if (tid < 64) b2s[tid] = b2[tid];

    const int wave = tid >> 5, lane = tid & 31;
    const int tile = blockIdx.x * 4 + wave;
    const bool active = (tile < numTiles);

    if (active && lane < 16) {
        long e = (long)tile * 16 + lane;
        float v[64], s1 = 0.f, s2 = 0.f;
        #pragma unroll
        for (int c = 0; c < 64; ++c) {
            float x = z[e * 64 + c] + acc[e * 64 + c];
            v[c] = x; s1 += x; s2 += x * x;
            acc[e * 64 + c] = x;   // stash residual
        }
        float mean = s1 * (1.f / 64.f);
        float var  = s2 * (1.f / 64.f) - mean * mean;
        float inv  = rsqrtf(var + 1e-5f);
        #pragma unroll
        for (int c = 0; c < 64; ++c) v[c] = (v[c] - mean) * inv * ln_g[c] + ln_b[c];
        #pragma unroll
        for (int ks = 0; ks < 2; ++ks) {
            v16h t0, t1;
            #pragma unroll
            for (int idx = 0; idx < 16; ++idx) {
                int kl = ((idx >> 3) * 16) + (idx & 7);
                t0[idx] = (_Float16)v[ks * 32 + kl];
                t1[idx] = (_Float16)v[ks * 32 + 8 + kl];
            }
            stFrag(Ssw + ((wave * 2 + ks) * 32 + lane) * 16, t0);
            stFrag(Ssw + ((wave * 2 + ks) * 32 + lane + 16) * 16, t1);
        }
    }
    __syncthreads();

    const int nl   = lane & 15;
    const int moff = (lane >= 16) ? 8 : 0;

    if (active) {
        v16h a0 = ldFrag(Ssw + ((wave * 2 + 0) * 32 + lane) * 16);
        v16h a1 = ldFrag(Ssw + ((wave * 2 + 1) * 32 + lane) * 16);
        #pragma unroll 4
        for (int nt = 0; nt < 16; ++nt) {
            v16h bf0 = ldFrag(W1sw + ((0 * 16 + nt) * 32 + lane) * 16);
            v16h bf1 = ldFrag(W1sw + ((1 * 16 + nt) * 32 + lane) * 16);
            v8f c = {};
            c = WMMA_F32_F16(a0, bf0, c);
            c = WMMA_F32_F16(a1, bf1, c);
            // scatter gelu(h) into swizzled A layout for GEMM2 (k dim = col n)
            int n    = nt * 16 + nl;
            int ksH  = n >> 5;
            int klH  = n & 31;
            int idxH = ((klH >= 16) ? 8 : 0) + (klH & 7);
            int lhi  = ((klH & 15) >= 8) ? 16 : 0;
            #pragma unroll
            for (int r = 0; r < 8; ++r) {
                float hv = c[r] + b1s[n];
                float g  = 0.5f * hv * (1.0f + erff(hv * 0.70710678118f));
                Hsw[((wave * 8 + ksH) * 32 + (r + moff + lhi)) * 16 + idxH] = (_Float16)g;
            }
        }
    }
    __syncthreads();
    if (!active) return;

    const long ebase = (long)tile * 16;
    #pragma unroll
    for (int nt = 0; nt < 4; ++nt) {
        v8f c = {};
        int n = nt * 16 + nl;
        #pragma unroll
        for (int ks = 0; ks < 8; ++ks) {
            v16h af = ldFrag(Hsw + ((wave * 8 + ks) * 32 + lane) * 16);
            v16h bf = ldFrag(W2sw + ((ks * 4 + nt) * 32 + lane) * 16);
            c = WMMA_F32_F16(af, bf, c);
        }
        #pragma unroll
        for (int r = 0; r < 8; ++r) {
            long e = ebase + r + moff;
            z[e * 64 + n] = acc[e * 64 + n] + c[r] + b2s[n];
        }
    }
}

// ---------------------------------------------------------------------------
// Host-side orchestration (input order per setup_inputs dict flattening).
// ---------------------------------------------------------------------------
extern "C" void kernel_launch(void* const* d_in, const int* in_sizes, int n_in,
                              void* d_out, int out_size, void* d_ws, size_t ws_size,
                              hipStream_t stream) {
    (void)in_sizes; (void)n_in; (void)out_size; (void)ws_size;
    constexpr long E = 30000L * 12;  // 360000 edges

    const float* edge_values = (const float*)d_in[0];
    const int*   nbr         = (const int*)d_in[1];
    auto F = [&](int i) { return (const float*)d_in[i]; };

    // workspace carve: a(f16,E*16) | b(f16,E*16) | go(f16,E*64) | acc(f32,E*64)
    char*     ws     = (char*)d_ws;
    _Float16* a_buf  = (_Float16*)ws;
    _Float16* b_buf  = (_Float16*)(ws + E * 32);
    _Float16* go_buf = (_Float16*)(ws + E * 64);
    float*    acc    = (float*)(ws + E * 192);
    float*    zz     = (float*)d_out;

    hipMemcpyAsync(zz, edge_values, E * 64 * sizeof(float),
                   hipMemcpyDeviceToDevice, stream);

    const int numTiles = (int)(E / 16);  // 22500
    dim3 gAB((numTiles + 7) / 8), bAB(256);
    dim3 gG(numTiles), bG(256);
    dim3 gF((numTiles + 3) / 4), bF(128);
    const size_t ffnSmem = 32768 + 32768 + 32768 + 8192 + 1024 + 256;

    for (int l = 0; l < 2; ++l) {
        for (int half = 0; half < 2; ++half) {   // 0 = row, 1 = col
            int base = 2 + half * 16;
            const float* ln_g  = F(base + 0)  + l * 64;
            const float* ln_b  = F(base + 1)  + l * 64;
            const float* la_W  = F(base + 2)  + l * 64 * 16;
            const float* la_b  = F(base + 3)  + l * 16;
            const float* ga_W  = F(base + 4)  + l * 64 * 16;
            const float* ga_b  = F(base + 5)  + l * 16;
            const float* lb_W  = F(base + 6)  + l * 64 * 16;
            const float* lb_b  = F(base + 7)  + l * 16;
            const float* gb_W  = F(base + 8)  + l * 64 * 16;
            const float* gb_b  = F(base + 9)  + l * 16;
            const float* go_W  = F(base + 10) + l * 64 * 64;
            const float* go_b  = F(base + 11) + l * 64;
            const float* lno_g = F(base + 12) + l * 16;
            const float* lno_b = F(base + 13) + l * 16;
            const float* lo_W  = F(base + 14) + l * 16 * 64;
            const float* lo_b  = F(base + 15) + l * 64;

            k_ab<<<gAB, bAB, 0, stream>>>(zz, ln_g, ln_b,
                                          ga_W, ga_b, la_W, la_b,
                                          gb_W, gb_b, lb_W, lb_b,
                                          go_W, go_b,
                                          a_buf, b_buf, go_buf, numTiles);
            k_gather<<<gG, bG, 0, stream>>>(a_buf, b_buf, go_buf, nbr,
                                            lno_g, lno_b, lo_W, lo_b,
                                            acc, half, (int)E);
        }
        const float* ffn_ln_g = F(34) + l * 64;
        const float* ffn_ln_b = F(35) + l * 64;
        const float* ffn_W1   = F(36) + l * 64 * 256;
        const float* ffn_b1   = F(37) + l * 256;
        const float* ffn_W2   = F(38) + l * 256 * 64;
        const float* ffn_b2   = F(39) + l * 64;
        k_ffn<<<gF, bF, ffnSmem, stream>>>(zz, acc, ffn_ln_g, ffn_ln_b,
                                           ffn_W1, ffn_b1, ffn_W2, ffn_b2, numTiles);
    }
}
